// ContrastiveLossWithViews_20796231647280
// MI455X (gfx1250) — matrix operand order repaired and checked by reading, not compiled, not run on hardware
//
#include <hip/hip_runtime.h>
#include <hip/hip_bf16.h>
#include <math.h>

// ---- types ----------------------------------------------------------------
typedef _Float16 half4  __attribute__((ext_vector_type(4)));
typedef _Float16 half8  __attribute__((ext_vector_type(8)));
typedef _Float16 half16 __attribute__((ext_vector_type(16)));
typedef float    f32x8  __attribute__((ext_vector_type(8)));

#define CL_B   4096   // batch per view
#define CL_D   1024   // feature dim
#define CL_R   8192   // 2*B rows
#define INV_T  10.0f  // 1/temperature
#define MBLK   32     // rows per block (2 WMMA M-tiles)

// ---------------------------------------------------------------------------
// Kernel 1: row L2-normalize z1||z2 -> f16 reps[8192][1024]
// ---------------------------------------------------------------------------
__global__ void __launch_bounds__(256) cl_normalize(
        const float* __restrict__ z1, const float* __restrict__ z2,
        _Float16* __restrict__ reps) {
    const int lane = threadIdx.x & 31;
    const int wave = threadIdx.x >> 5;
    const int row  = blockIdx.x * 8 + wave;
    const float* src = (row < CL_B) ? (z1 + (size_t)row * CL_D)
                                    : (z2 + (size_t)(row - CL_B) * CL_D);
    float ss = 0.0f;
    float4 v[8];
#pragma unroll
    for (int w = 0; w < 8; ++w) {
        v[w] = *(const float4*)(src + w * 128 + lane * 4);
        ss += v[w].x * v[w].x + v[w].y * v[w].y + v[w].z * v[w].z + v[w].w * v[w].w;
    }
    ss += __shfl_xor(ss, 1);
    ss += __shfl_xor(ss, 2);
    ss += __shfl_xor(ss, 4);
    ss += __shfl_xor(ss, 8);
    ss += __shfl_xor(ss, 16);
    const float scale = 1.0f / fmaxf(sqrtf(ss), 1e-12f);
    _Float16* dst = reps + (size_t)row * CL_D;
#pragma unroll
    for (int w = 0; w < 8; ++w) {
        half4 h;
        h.x = (_Float16)(v[w].x * scale);
        h.y = (_Float16)(v[w].y * scale);
        h.z = (_Float16)(v[w].z * scale);
        h.w = (_Float16)(v[w].w * scale);
        *(half4*)(dst + w * 128 + lane * 4) = h;
    }
}

// ---- fragment load helpers ------------------------------------------------
// A operand (M x K, contiguous K layout): low lanes K{0..15},
// high lanes K{16..31} of row nlo within the M-tile.
__device__ __forceinline__ half16 load_a_frag(const _Float16* lds_a, int mt,
                                              int nlo, int half, int k0) {
    const _Float16* ab = lds_a + (mt * 16 + nlo) * CL_D + k0 + 16 * half;
    const half8 alo = *(const half8*)(ab);
    const half8 ahi = *(const half8*)(ab + 8);
    return __builtin_shufflevector(alo, ahi,
        0, 1, 2, 3, 4, 5, 6, 7, 8, 9, 10, 11, 12, 13, 14, 15);
}
// B operand (K x N, column-per-lane): lane nlo holds contiguous K of row
// (colBase + t*16 + nlo); lanes 16-31 take the K+16 slice.
__device__ __forceinline__ half16 load_b_frag(const _Float16* bbase0, int t,
                                              int k0) {
    return *(const half16*)(bbase0 + (size_t)t * 16 * CL_D + k0);
}

// ---------------------------------------------------------------------------
// Kernel 2: fused sim-GEMM (WMMA f16->f32) + masked logsumexp per row.
// Block = 32 rows x all 8192 cols; 8 waves; each wave computes a 2(M)x4(N)
// accumulator grid per column group. K loop is manually ping-pong
// software-pipelined so B loads run under the previous buffer's 8 WMMAs.
// ---------------------------------------------------------------------------
__global__ void __launch_bounds__(256, 1) cl_gemm_lse(
        const _Float16* __restrict__ reps, float* __restrict__ partials) {
    __shared__ _Float16 lds_a[MBLK * CL_D];    // 64 KB row tile
    __shared__ float lds_partial[8][MBLK];
    __shared__ float lds_pos[MBLK];
    __shared__ float lds_c[MBLK];

    const int tid  = threadIdx.x;
    const int lane = tid & 31;
    const int wave = tid >> 5;
    const int half = lane >> 4;
    const int nlo  = lane & 15;
    const int rowBase = blockIdx.x * MBLK;

    // stage A tile (32 rows, contiguous) into LDS
    {
        const _Float16* src = reps + (size_t)rowBase * CL_D;
#pragma unroll
        for (int i = tid; i < MBLK * CL_D / 8; i += 256)
            *(half8*)(lds_a + i * 8) = *(const half8*)(src + i * 8);
    }
    if (tid < MBLK) lds_pos[tid] = 0.0f;
    __syncthreads();

    float accL[2][8] = {};

    // 512 column tiles -> 128 groups of 4; waves interleave over groups
    for (int g = wave; g < 128; g += 8) {
        const int colBase = g * 64;
        f32x8 acc[2][4] = {};
        const _Float16* bbase0 = reps + (size_t)(colBase + nlo) * CL_D + 16 * half;

        // ---- pipelined K loop: ping (a0/b0) / pong (a1/b1) ----
        half16 a0[2], a1[2], b0[4], b1[4];
#pragma unroll
        for (int mt = 0; mt < 2; ++mt) a0[mt] = load_a_frag(lds_a, mt, nlo, half, 0);
#pragma unroll
        for (int t = 0; t < 4; ++t) b0[t] = load_b_frag(bbase0, t, 0);

#pragma unroll 1
        for (int k0 = 0; k0 < CL_D; k0 += 64) {
            __builtin_prefetch(bbase0 + k0 + 64, 0, 1);
            // prefetch pong buffer @ k0+32
#pragma unroll
            for (int mt = 0; mt < 2; ++mt)
                a1[mt] = load_a_frag(lds_a, mt, nlo, half, k0 + 32);
#pragma unroll
            for (int t = 0; t < 4; ++t) b1[t] = load_b_frag(bbase0, t, k0 + 32);
            // consume ping buffer @ k0
#pragma unroll
            for (int t = 0; t < 4; ++t)
#pragma unroll
                for (int mt = 0; mt < 2; ++mt)
                    acc[mt][t] = __builtin_amdgcn_wmma_f32_16x16x32_f16(
                        false, a0[mt], false, b0[t], (short)0, acc[mt][t],
                        false, false);
            // prefetch next ping buffer @ k0+64 (wraps dead on last iter)
            const int k2 = (k0 + 64) & (CL_D - 1);
#pragma unroll
            for (int mt = 0; mt < 2; ++mt)
                a0[mt] = load_a_frag(lds_a, mt, nlo, half, k2);
#pragma unroll
            for (int t = 0; t < 4; ++t) b0[t] = load_b_frag(bbase0, t, k2);
            // consume pong buffer @ k0+32
#pragma unroll
            for (int t = 0; t < 4; ++t)
#pragma unroll
                for (int mt = 0; mt < 2; ++mt)
                    acc[mt][t] = __builtin_amdgcn_wmma_f32_16x16x32_f16(
                        false, a1[mt], false, b1[t], (short)0, acc[mt][t],
                        false, false);
        }

        // tile epilogue: scale by 1/T, mask diagonal, extract positive,
        // accumulate exp(sim - 10)   (sim <= 10 always)
#pragma unroll
        for (int t = 0; t < 4; ++t) {
            const int gc = colBase + t * 16 + nlo;
#pragma unroll
            for (int mt = 0; mt < 2; ++mt) {
#pragma unroll
                for (int v = 0; v < 8; ++v) {
                    const int m  = mt * 16 + v + 8 * half;
                    const int gi = rowBase + m;
                    const float s = acc[mt][t][v] * INV_T;
                    const float e = (gc == gi) ? 0.0f : __expf(s - 10.0f);
                    accL[mt][v] += e;
                    if (gc == (gi ^ CL_B)) lds_pos[m] = s;
                }
            }
        }
    }

    // reduce row sums across the 16 lanes of each half (fixed order)
#pragma unroll
    for (int mt = 0; mt < 2; ++mt) {
#pragma unroll
        for (int v = 0; v < 8; ++v) {
            float s = accL[mt][v];
            s += __shfl_xor(s, 1);
            s += __shfl_xor(s, 2);
            s += __shfl_xor(s, 4);
            s += __shfl_xor(s, 8);
            if (nlo == 0) lds_partial[wave][mt * 16 + v + 8 * half] = s;
        }
    }
    __syncthreads();

    if (tid < MBLK) {
        float S = 0.0f;
#pragma unroll
        for (int w = 0; w < 8; ++w) S += lds_partial[w][tid];
        const float p = lds_pos[tid];
        lds_c[tid] = 10.0f - p + logf(__expf(p - 10.0f) + S);   // lse - pos
    }
    __syncthreads();
    if (tid == 0) {
        float s = 0.0f;
#pragma unroll
        for (int m = 0; m < MBLK; ++m) s += lds_c[m];
        partials[blockIdx.x] = s;
    }
}

// ---------------------------------------------------------------------------
// Kernel 3: deterministic tree reduction of 256 block partials -> mean
// ---------------------------------------------------------------------------
__global__ void __launch_bounds__(256) cl_finalize(
        const float* __restrict__ partials, float* __restrict__ out) {
    __shared__ float red[256];
    const int tid = threadIdx.x;
    red[tid] = partials[tid];
    __syncthreads();
    for (int s = 128; s > 0; s >>= 1) {
        if (tid < s) red[tid] += red[tid + s];
        __syncthreads();
    }
    if (tid == 0) out[0] = red[0] / (float)CL_R;
}

// ---------------------------------------------------------------------------
extern "C" void kernel_launch(void* const* d_in, const int* in_sizes, int n_in,
                              void* d_out, int out_size, void* d_ws, size_t ws_size,
                              hipStream_t stream) {
    const float* z1 = (const float*)d_in[0];
    const float* z2 = (const float*)d_in[1];
    float* out = (float*)d_out;

    _Float16* reps = (_Float16*)d_ws;                              // 16 MB
    float* partials = (float*)((char*)d_ws + (size_t)CL_R * CL_D * 2);

    cl_normalize<<<CL_R / 8, 256, 0, stream>>>(z1, z2, reps);
    cl_gemm_lse<<<CL_R / MBLK, 256, 0, stream>>>(reps, partials);
    cl_finalize<<<1, 256, 0, stream>>>(partials, out);
}
